// TreeNodeEmbeddings_23055384445128
// MI455X (gfx1250) — matrix operand order repaired and checked by reading, not compiled, not run on hardware
//
#include <hip/hip_runtime.h>

typedef __attribute__((ext_vector_type(16))) __bf16 v16bf;
typedef __attribute__((ext_vector_type(8)))  __bf16 v8bf;
typedef __attribute__((ext_vector_type(8)))  float  v8f;

#define RANK 64
#define CHUNKS 2
#define ROWS_PER_BLOCK 64

__global__ __launch_bounds__(256) void tree_mlp_kernel(
    const float* __restrict__ weight,
    const float* __restrict__ W1,
    const float* __restrict__ b1,
    const float* __restrict__ W2,
    const float* __restrict__ b2,
    const int*   __restrict__ nodeIdx,
    const int*   __restrict__ leaf_mask,
    float*       __restrict__ out,
    int batch)
{
    __shared__ __align__(16) float  sE[ROWS_PER_BLOCK * RANK];   // gathered f32 embeddings, later reused as f32 output staging
    __shared__ __align__(32) __bf16 sW1T[RANK * RANK];           // W1 transposed [n][k], bf16
    __shared__ __align__(32) __bf16 sW2T[RANK * RANK];           // W2 transposed [n][k], bf16
    __shared__ __align__(32) __bf16 sH[ROWS_PER_BLOCK * RANK];   // layer-1 activations, bf16
    __shared__ float sB1[RANK];
    __shared__ float sB2[RANK];
    __shared__ int   sIdx[ROWS_PER_BLOCK];
    __shared__ int   sLeaf[ROWS_PER_BLOCK];

    const int tid = threadIdx.x;
    const int c   = blockIdx.y;                 // chunk
    const int b0  = blockIdx.x * ROWS_PER_BLOCK;

    // ---- stage node indices ----
    if (tid < ROWS_PER_BLOCK) sIdx[tid] = nodeIdx[b0 + tid];
    __syncthreads();

    // ---- gather embeddings for this chunk: async Global -> LDS (B128), no VGPR staging ----
    {
        const unsigned ldsBase = (unsigned)(uintptr_t)(&sE[0]);
#pragma unroll
        for (int t = 0; t < 4; t++) {
            int i   = tid + t * 256;                              // 1024 float4 transfers
            int row = i >> 4;                                     // 16 float4 per row
            int q   = i & 15;
            const float* src =
                weight + (size_t)sIdx[row] * (RANK * CHUNKS) + c * RANK + q * 4;
            unsigned ldsOff = ldsBase + (unsigned)(i * 16);
            asm volatile("global_load_async_to_lds_b128 %0, %1, off"
                         :: "v"(ldsOff), "v"(src) : "memory");
        }
    }

    // ---- weights -> bf16, transposed to [n][k] so B-fragments are contiguous (overlaps async gather) ----
    {
        const float* w1c = W1 + (size_t)c * RANK * RANK;
        const float* w2c = W2 + (size_t)c * RANK * RANK;
        for (int i = tid; i < RANK * RANK; i += 256) {
            int r = i >> 6;        // K (input dim)
            int k = i & 63;        // N (output dim)
            sW1T[k * RANK + r] = (__bf16)w1c[i];
            sW2T[k * RANK + r] = (__bf16)w2c[i];
        }
    }
    if (tid < RANK) { sB1[tid] = b1[c * RANK + tid]; sB2[tid] = b2[c * RANK + tid]; }
    if (tid < ROWS_PER_BLOCK) sLeaf[tid] = leaf_mask[sIdx[tid]];

    asm volatile("s_wait_asynccnt 0x0" ::: "memory");   // my async gather landed in LDS
    __syncthreads();                                     // everyone's gather visible

    const int wave   = tid >> 5;
    const int lane   = tid & 31;
    const int hlf    = lane >> 4;      // lane half (K/M sub-block select)
    const int lcol   = lane & 15;
    const int rowblk = wave & 3;       // which 16-row tile
    const int nhalf  = wave >> 2;      // which 32-col half
    const int m0     = rowblk * 16;

    // ================= layer 1: h = relu(e @ W1 + b1) =================
    // A-fragments (16x32 bf16, ISA layout): per lane two contiguous 8-wide K runs
    v16bf a0, a1;
    {
        const float* eRow = &sE[(m0 + lcol) * RANK];
#pragma unroll
        for (int ks = 0; ks < 2; ks++) {
            int base0 = ks * 32 + hlf * 8;
            int base1 = base0 + 16;
            v16bf a;
#pragma unroll
            for (int t = 0; t < 8; t++) a[t]     = (__bf16)eRow[base0 + t];
#pragma unroll
            for (int t = 0; t < 8; t++) a[8 + t] = (__bf16)eRow[base1 + t];
            if (ks == 0) a0 = a; else a1 = a;
        }
    }

#pragma unroll
    for (int nt = 0; nt < 2; nt++) {
        int n   = nhalf * 2 + nt;
        int col = n * 16 + lcol;
        v8f acc;
        float bv = sB1[col];
#pragma unroll
        for (int j = 0; j < 8; j++) acc[j] = bv;
        // B-fragment (32x16 bf16): K = hlf*8 + i (i<8), 16 + hlf*8 + (i-8) (i>=8), from sW1T[col][*]
        v16bf bb0 = __builtin_shufflevector(*(const v8bf*)&sW1T[col * RANK + 0 * 32 + hlf * 8], *(const v8bf*)&sW1T[col * RANK + 0 * 32 + 16 + hlf * 8], 0,1,2,3,4,5,6,7,8,9,10,11,12,13,14,15);
        v16bf bb1 = __builtin_shufflevector(*(const v8bf*)&sW1T[col * RANK + 1 * 32 + hlf * 8], *(const v8bf*)&sW1T[col * RANK + 1 * 32 + 16 + hlf * 8], 0,1,2,3,4,5,6,7,8,9,10,11,12,13,14,15);
        acc = __builtin_amdgcn_wmma_f32_16x16x32_bf16(false, a0, false, bb0, (short)0, acc, false, false);
        acc = __builtin_amdgcn_wmma_f32_16x16x32_bf16(false, a1, false, bb1, (short)0, acc, false, false);
#pragma unroll
        for (int j = 0; j < 8; j++) {
            float hv = acc[j] > 0.f ? acc[j] : 0.f;             // relu
            sH[(m0 + hlf * 8 + j) * RANK + col] = (__bf16)hv;   // C-layout -> row-major via LDS
        }
    }
    __syncthreads();   // column-half waves exchange their h columns

    // ================= layer 2: out = h @ W2 + b2 =================
    {
        const __bf16* hRow = &sH[(m0 + lcol) * RANK];
#pragma unroll
        for (int ks = 0; ks < 2; ks++) {
            int base0 = ks * 32 + hlf * 8;
            v8bf lo = *(const v8bf*)&hRow[base0];
            v8bf hi = *(const v8bf*)&hRow[base0 + 16];
            v16bf a;
#pragma unroll
            for (int t = 0; t < 8; t++) { a[t] = lo[t]; a[8 + t] = hi[t]; }
            if (ks == 0) a0 = a; else a1 = a;
        }
    }

#pragma unroll
    for (int nt = 0; nt < 2; nt++) {
        int n   = nhalf * 2 + nt;
        int col = n * 16 + lcol;
        v8f acc;
        float bv = sB2[col];
#pragma unroll
        for (int j = 0; j < 8; j++) acc[j] = bv;
        v16bf bb0 = __builtin_shufflevector(*(const v8bf*)&sW2T[col * RANK + 0 * 32 + hlf * 8], *(const v8bf*)&sW2T[col * RANK + 0 * 32 + 16 + hlf * 8], 0,1,2,3,4,5,6,7,8,9,10,11,12,13,14,15);
        v16bf bb1 = __builtin_shufflevector(*(const v8bf*)&sW2T[col * RANK + 1 * 32 + hlf * 8], *(const v8bf*)&sW2T[col * RANK + 1 * 32 + 16 + hlf * 8], 0,1,2,3,4,5,6,7,8,9,10,11,12,13,14,15);
        acc = __builtin_amdgcn_wmma_f32_16x16x32_bf16(false, a0, false, bb0, (short)0, acc, false, false);
        acc = __builtin_amdgcn_wmma_f32_16x16x32_bf16(false, a1, false, bb1, (short)0, acc, false, false);
        // leaf/non-leaf select; stage f32 result back into sE in place
        // (each (row,col) element is read+written by exactly the one lane that owns it)
#pragma unroll
        for (int j = 0; j < 8; j++) {
            int grow = m0 + hlf * 8 + j;                  // row within block
            float ev = sE[grow * RANK + col];             // exact f32 passthrough for leaves
            float o  = (sLeaf[grow] == 0) ? acc[j] : ev;
            sE[grow * RANK + col] = o;
        }
    }
    __syncthreads();   // all staged results visible / DS ops drained before async store

    // ---- stream the block's contiguous 16KB output span: async LDS -> Global (B128) ----
    {
        const unsigned ldsBase = (unsigned)(uintptr_t)(&sE[0]);
        const float4*  outV    = (const float4*)(out + (size_t)c * (size_t)batch * RANK
                                                     + (size_t)b0 * RANK);
#pragma unroll
        for (int t = 0; t < 4; t++) {
            int i = tid + t * 256;                         // 1024 float4 stores
            const float4* dst = outV + i;
            unsigned ldsOff = ldsBase + (unsigned)(i * 16);
            asm volatile("global_store_async_from_lds_b128 %0, %1, off"
                         :: "v"(dst), "v"(ldsOff) : "memory");
        }
    }
    asm volatile("s_wait_asynccnt 0x0" ::: "memory");      // drain before wave exit (S_ENDPGM would also wait-idle)
}

extern "C" void kernel_launch(void* const* d_in, const int* in_sizes, int n_in,
                              void* d_out, int out_size, void* d_ws, size_t ws_size,
                              hipStream_t stream) {
    (void)n_in; (void)out_size; (void)d_ws; (void)ws_size;
    const float* weight   = (const float*)d_in[0];
    const float* W1       = (const float*)d_in[1];
    const float* b1       = (const float*)d_in[2];
    const float* W2       = (const float*)d_in[3];
    const float* b2       = (const float*)d_in[4];
    const int*   nodeIdx  = (const int*)d_in[5];
    const int*   leafMask = (const int*)d_in[6];
    float*       out      = (float*)d_out;

    int batch = in_sizes[5];                       // 524288
    dim3 grid(batch / ROWS_PER_BLOCK, CHUNKS);     // (8192, 2)
    tree_mlp_kernel<<<grid, 256, 0, stream>>>(weight, W1, b1, W2, b2,
                                              nodeIdx, leafMask, out, batch);
}
